// Attention_2345052144007
// MI455X (gfx1250) — matrix operand order repaired and checked by reading, not compiled, not run on hardware
//
#include <hip/hip_runtime.h>

typedef __bf16 bf16;
typedef __attribute__((ext_vector_type(8)))  __bf16 v8bf;
typedef __attribute__((ext_vector_type(16))) __bf16 v16bf;
typedef __attribute__((ext_vector_type(8)))  float  v8f;

#define DIM   2048
#define SEQ   2048
#define BSZ   2
#define NH    32
#define NKV   8
#define HD    64
#define MTOT  (BSZ * SEQ)      // 4096
#define NQKV  3072             // (NH + 2*NKV) * HD

__device__ __forceinline__ v8f wmma_bf16(v16bf a, v16bf b, v8f c) {
  // v_wmma_f32_16x16x32_bf16  D = A(16x32) * B(32x16) + C
  return __builtin_amdgcn_wmma_f32_16x16x32_bf16(false, a, false, b, (short)0, c,
                                                 false, false);
}

// ---- DPP16 butterfly reductions within each 16-lane row (wave32: 2 rows).
// quad_perm xor1 = 0xB1, quad_perm xor2 = 0x4E,
// row_half_mirror = 0x141 (combines quads within 8),
// row_mirror      = 0x140 (combines 8-groups within 16).
template <int CTRL>
__device__ __forceinline__ float dpp_mov(float v) {
  return __int_as_float(__builtin_amdgcn_update_dpp(
      0, __float_as_int(v), CTRL, 0xf, 0xf, true));
}
__device__ __forceinline__ float red_max16(float v) {
  v = fmaxf(v, dpp_mov<0xB1>(v));
  v = fmaxf(v, dpp_mov<0x4E>(v));
  v = fmaxf(v, dpp_mov<0x141>(v));
  v = fmaxf(v, dpp_mov<0x140>(v));
  return v;
}
__device__ __forceinline__ float red_sum16(float v) {
  v += dpp_mov<0xB1>(v);
  v += dpp_mov<0x4E>(v);
  v += dpp_mov<0x141>(v);
  v += dpp_mov<0x140>(v);
  return v;
}

// ---------------------------------------------------------------- casts
__global__ void cast_f32_bf16(const float* __restrict__ in, bf16* __restrict__ out,
                              int n) {
  int i = blockIdx.x * blockDim.x + threadIdx.x;
  int stride = gridDim.x * blockDim.x;
  for (; i < n; i += stride) out[i] = (bf16)in[i];
}

// ------------------------------------------------- generic GEMM: A @ W^T
// A: [M,K] bf16 row-major.  W: [N,K] bf16 row-major.  out: [M,N] OutT.
// Block = 256 threads = 8 waves; wave w owns a 32x64 output tile at
// rows blockIdx.x*32, cols blockIdx.y*512 + w*64.
// Per 32-K step: 2 A-frags + 4 B-frags feed 8 WMMAs.
template <typename OutT>
__global__ void __launch_bounds__(256)
gemm_xwT(const bf16* __restrict__ A, const bf16* __restrict__ W,
         OutT* __restrict__ out, int K, int N) {
  const int lane = threadIdx.x & 31;
  const int wave = threadIdx.x >> 5;
  const int half = lane >> 4;     // 0 or 1 (lane group)
  const int l16  = lane & 15;
  const int mbase = blockIdx.x * 32;
  const int nb    = blockIdx.y * 512 + wave * 64;

  const bf16* Arow0 = A + (size_t)(mbase + l16) * K;
  const bf16* Arow1 = Arow0 + (size_t)16 * K;

  v8f acc00 = {}, acc01 = {}, acc02 = {}, acc03 = {};
  v8f acc10 = {}, acc11 = {}, acc12 = {}, acc13 = {};

  for (int k0 = 0; k0 < K; k0 += 32) {
    // A fragment 16x32: lane holds row l16; k = i + 16*half
    v8bf a0lo = *(const v8bf*)(Arow0 + k0 + 16 * half);
    v8bf a0hi = *(const v8bf*)(Arow0 + k0 + 8 + 16 * half);
    v8bf a1lo = *(const v8bf*)(Arow1 + k0 + 16 * half);
    v8bf a1hi = *(const v8bf*)(Arow1 + k0 + 8 + 16 * half);
    v16bf af0, af1;
#pragma unroll
    for (int i = 0; i < 8; ++i) {
      af0[i] = a0lo[i]; af0[8 + i] = a0hi[i];
      af1[i] = a1lo[i]; af1[8 + i] = a1hi[i];
    }

    // B fragment 32x16: lane holds column n=l16 (a row of W); k = i + 16*half
    const bf16* wbase = W + (size_t)(nb + l16) * K + k0 + 16 * half;
    v16bf b0 = *(const v16bf*)(wbase);
    v16bf b1 = *(const v16bf*)(wbase + (size_t)16 * K);
    v16bf b2 = *(const v16bf*)(wbase + (size_t)32 * K);
    v16bf b3 = *(const v16bf*)(wbase + (size_t)48 * K);

    acc00 = wmma_bf16(af0, b0, acc00);
    acc01 = wmma_bf16(af0, b1, acc01);
    acc02 = wmma_bf16(af0, b2, acc02);
    acc03 = wmma_bf16(af0, b3, acc03);
    acc10 = wmma_bf16(af1, b0, acc10);
    acc11 = wmma_bf16(af1, b1, acc11);
    acc12 = wmma_bf16(af1, b2, acc12);
    acc13 = wmma_bf16(af1, b3, acc13);
  }

  // C layout: VGPR r holds row (tile_m + r + 8*half), column nb + t*16 + l16
#pragma unroll
  for (int r = 0; r < 8; ++r) {
    size_t m0 = (size_t)(mbase + r + 8 * half);
    size_t m1 = m0 + 16;
    out[m0 * N + nb + 0  + l16] = (OutT)acc00[r];
    out[m0 * N + nb + 16 + l16] = (OutT)acc01[r];
    out[m0 * N + nb + 32 + l16] = (OutT)acc02[r];
    out[m0 * N + nb + 48 + l16] = (OutT)acc03[r];
    out[m1 * N + nb + 0  + l16] = (OutT)acc10[r];
    out[m1 * N + nb + 16 + l16] = (OutT)acc11[r];
    out[m1 * N + nb + 32 + l16] = (OutT)acc12[r];
    out[m1 * N + nb + 48 + l16] = (OutT)acc13[r];
  }
}

// ---------------------------------------------- RoPE + layout scatter
// qkv: [b*s, 3072] bf16.  fc: [SEQ, 32, 2] f32 (cos,sin).
// qb:  [b, h, s, 64]   (A-side, contiguous d)
// kb:  [b, kv, s, 64]  (B-side for Q@K^T, contiguous d)
// vtb: [b, kv, 64, s]  (B-side for P@V, contiguous s)
__global__ void rope_scatter(const bf16* __restrict__ qkv, const float* __restrict__ fc,
                             bf16* __restrict__ qb, bf16* __restrict__ kb,
                             bf16* __restrict__ vtb) {
  const int bs = blockIdx.x;
  const int b = bs / SEQ, s = bs % SEQ;
  const bf16* row = qkv + (size_t)bs * NQKV;
  for (int p = threadIdx.x; p < NQKV / 2; p += blockDim.x) {
    const int dd = 2 * p;
    float x0 = (float)row[dd], x1 = (float)row[dd + 1];
    const int hd = dd & 63;
    float y0 = x0, y1 = x1;
    if (dd < DIM + NKV * HD) {  // rotary applies to q and k only
      const int i = hd >> 1;
      const float c  = fc[((size_t)s * 32 + i) * 2 + 0];
      const float sn = fc[((size_t)s * 32 + i) * 2 + 1];
      y0 = x0 * c - x1 * sn;
      y1 = x1 * c + x0 * sn;
    }
    if (dd < DIM) {
      const int h = dd >> 6;
      size_t o = ((size_t)(b * NH + h) * SEQ + s) * HD + hd;
      qb[o] = (bf16)y0; qb[o + 1] = (bf16)y1;
    } else if (dd < DIM + NKV * HD) {
      const int kv = (dd - DIM) >> 6;
      size_t o = ((size_t)(b * NKV + kv) * SEQ + s) * HD + hd;
      kb[o] = (bf16)y0; kb[o + 1] = (bf16)y1;
    } else {
      const int kv = (dd - DIM - NKV * HD) >> 6;
      size_t o = ((size_t)(b * NKV + kv) * HD + hd) * SEQ + s;
      vtb[o] = (bf16)y0; vtb[o + SEQ] = (bf16)y1;  // rows hd, hd+1 of V^T
    }
  }
}

// -------------------------------------------------- flash attention
// One wave per 16-query tile; Q fragments resident; 32-key chunks:
// 4 score WMMAs + online softmax (DPP16 row reductions, no LDS) +
// LDS transpose of P + 4 P@V WMMAs.  Key loop split: unmasked main
// chunks + one masked diagonal chunk.
__global__ void __launch_bounds__(128)
flash_attn(const bf16* __restrict__ qb, const bf16* __restrict__ kb,
           const bf16* __restrict__ vtb, bf16* __restrict__ attn_out) {
  __shared__ __align__(32) bf16 pls[4][16 * 32];
  const int lane = threadIdx.x & 31;
  const int wave = threadIdx.x >> 5;
  const int half = lane >> 4;
  const int l16  = lane & 15;
  const int h = blockIdx.y;
  const int b = blockIdx.z;
  const int qbase = (blockIdx.x * 4 + wave) * 16;
  const int bh  = b * NH + h;
  const int bkv = b * NKV + (h >> 2);  // GQA: 4 Q heads per KV head

  // resident Q: 16x64 -> two 16x32 A fragments
  const bf16* qrow = qb + ((size_t)bh * SEQ + qbase + l16) * HD;
  v16bf qa0, qa1;
  {
    v8bf lo0 = *(const v8bf*)(qrow + 16 * half);
    v8bf hi0 = *(const v8bf*)(qrow + 8 + 16 * half);
    v8bf lo1 = *(const v8bf*)(qrow + 32 + 16 * half);
    v8bf hi1 = *(const v8bf*)(qrow + 40 + 16 * half);
#pragma unroll
    for (int i = 0; i < 8; ++i) {
      qa0[i] = lo0[i]; qa0[8 + i] = hi0[i];
      qa1[i] = lo1[i]; qa1[8 + i] = hi1[i];
    }
  }

  float mrun[8], lrun[8];
  v8f o0 = {}, o1 = {}, o2 = {}, o3 = {};
#pragma unroll
  for (int r = 0; r < 8; ++r) { mrun[r] = -1e30f; lrun[r] = 0.0f; }

  bf16* pl = &pls[wave][0];

  // one 32-key chunk; `masked` is always a compile-time constant at call sites
  auto chunk = [&](int jb, bool masked) __attribute__((always_inline)) {
    // ---- scores for two 16-key tiles
    float c0[8], c1[8];
#pragma unroll
    for (int t = 0; t < 2; ++t) {
      const bf16* krow = kb + ((size_t)bkv * SEQ + jb + t * 16 + l16) * HD;
      v8f acc = {};
      v16bf kf0 = *(const v16bf*)(krow + 16 * half);
      v16bf kf1 = *(const v16bf*)(krow + 32 + 16 * half);
      acc = wmma_bf16(qa0, kf0, acc);
      acc = wmma_bf16(qa1, kf1, acc);
      float* ct = t ? c1 : c0;
#pragma unroll
      for (int r = 0; r < 8; ++r) {
        float v = acc[r] * 0.125f;  // 1/sqrt(64)
        if (masked) {
          const int q  = qbase + r + 8 * half;
          const int kk = jb + t * 16 + l16;
          v = (kk <= q) ? v : -1e30f;
        }
        ct[r] = v;
      }
    }
    // ---- online softmax (row stats live within each 16-lane row; DPP16)
    float alpha[8];
#pragma unroll
    for (int r = 0; r < 8; ++r) {
      const float rm = red_max16(fmaxf(c0[r], c1[r]));
      const float mn = fmaxf(mrun[r], rm);
      alpha[r] = __expf(mrun[r] - mn);
      const float p0 = __expf(c0[r] - mn);
      const float p1 = __expf(c1[r] - mn);
      c0[r] = p0; c1[r] = p1;
      const float rs = red_sum16(p0 + p1);
      lrun[r] = lrun[r] * alpha[r] + rs;
      mrun[r] = mn;
    }
    // ---- transpose P (C layout -> A layout) via LDS
#pragma unroll
    for (int r = 0; r < 8; ++r) {
      const int m = r + 8 * half;
      pl[m * 32 + l16]      = (bf16)c0[r];
      pl[m * 32 + 16 + l16] = (bf16)c1[r];
    }
    asm volatile("s_wait_dscnt 0" ::: "memory");
    v16bf pa;
    {
      const bf16* prow = pl + l16 * 32;
      v8bf lo = *(const v8bf*)(prow + 16 * half);
      v8bf hi = *(const v8bf*)(prow + 8 + 16 * half);
#pragma unroll
      for (int i = 0; i < 8; ++i) { pa[i] = lo[i]; pa[8 + i] = hi[i]; }
    }
    // ---- rescale O, accumulate P@V (V^T rows contiguous along s)
#pragma unroll
    for (int r = 0; r < 8; ++r) {
      o0[r] *= alpha[r]; o1[r] *= alpha[r]; o2[r] *= alpha[r]; o3[r] *= alpha[r];
    }
    const bf16* vrow = vtb + ((size_t)bkv * HD + l16) * SEQ + jb + 16 * half;
    v16bf vb0 = *(const v16bf*)(vrow);
    v16bf vb1 = *(const v16bf*)(vrow + (size_t)16 * SEQ);
    v16bf vb2 = *(const v16bf*)(vrow + (size_t)32 * SEQ);
    v16bf vb3 = *(const v16bf*)(vrow + (size_t)48 * SEQ);
    o0 = wmma_bf16(pa, vb0, o0);
    o1 = wmma_bf16(pa, vb1, o1);
    o2 = wmma_bf16(pa, vb2, o2);
    o3 = wmma_bf16(pa, vb3, o3);
  };

  // main loop: every key in [jb, jb+31] is < qbase <= q  -> no masking
  int jb = 0;
  for (; jb + 32 <= qbase; jb += 32) chunk(jb, false);
  // exactly one trailing chunk contains the diagonal -> masked
  chunk(jb, true);

  // ---- normalize + store [b, s, h*64 + d]
#pragma unroll
  for (int r = 0; r < 8; ++r) {
    const int q = qbase + r + 8 * half;
    const float inv = 1.0f / lrun[r];
    size_t base = ((size_t)(b * SEQ + q)) * DIM + h * HD;
    attn_out[base + l16]      = (bf16)(o0[r] * inv);
    attn_out[base + 16 + l16] = (bf16)(o1[r] * inv);
    attn_out[base + 32 + l16] = (bf16)(o2[r] * inv);
    attn_out[base + 48 + l16] = (bf16)(o3[r] * inv);
  }
}

// ------------------------------------------------------------ launcher
extern "C" void kernel_launch(void* const* d_in, const int* in_sizes, int n_in,
                              void* d_out, int out_size, void* d_ws, size_t ws_size,
                              hipStream_t stream) {
  (void)in_sizes; (void)n_in; (void)out_size; (void)ws_size;
  const float* x    = (const float*)d_in[0];
  const float* fc   = (const float*)d_in[1];
  /* d_in[2] = mask: causal, computed analytically */
  const float* wqkv = (const float*)d_in[3];
  const float* wo   = (const float*)d_in[4];
  float* out = (float*)d_out;

  char* ws = (char*)d_ws;
  size_t off = 0;
  bf16* xb    = (bf16*)(ws + off); off += (size_t)MTOT * DIM * 2;        // 16 MiB
  bf16* wqkvb = (bf16*)(ws + off); off += (size_t)NQKV * DIM * 2;        // 12 MiB
  bf16* wob   = (bf16*)(ws + off); off += (size_t)DIM * DIM * 2;         //  8 MiB
  size_t qkv_off = off;
  bf16* qkvb  = (bf16*)(ws + off); off += (size_t)MTOT * NQKV * 2;       // 24 MiB
  bf16* qb    = (bf16*)(ws + off); off += (size_t)BSZ * NH * SEQ * HD * 2;   // 16 MiB
  bf16* kbuf  = (bf16*)(ws + off); off += (size_t)BSZ * NKV * SEQ * HD * 2;  //  4 MiB
  bf16* vtb   = (bf16*)(ws + off); off += (size_t)BSZ * NKV * HD * SEQ * 2;  //  4 MiB
  bf16* aout  = (bf16*)(ws + qkv_off);  // reuse dead qkv region

  cast_f32_bf16<<<4096, 256, 0, stream>>>(x, xb, MTOT * DIM);
  cast_f32_bf16<<<4096, 256, 0, stream>>>(wqkv, wqkvb, NQKV * DIM);
  cast_f32_bf16<<<4096, 256, 0, stream>>>(wo, wob, DIM * DIM);

  gemm_xwT<bf16><<<dim3(MTOT / 32, NQKV / 512), 256, 0, stream>>>(xb, wqkvb, qkvb,
                                                                  DIM, NQKV);
  rope_scatter<<<MTOT, 256, 0, stream>>>(qkvb, fc, qb, kbuf, vtb);
  flash_attn<<<dim3(SEQ / 64, NH, BSZ), 128, 0, stream>>>(qb, kbuf, vtb, aout);
  gemm_xwT<float><<<dim3(MTOT / 32, DIM / 512), 256, 0, stream>>>(aout, wob, out,
                                                                  DIM, DIM);
}
